// ExplaiNN_10788957848055
// MI455X (gfx1250) — matrix operand order repaired and checked by reading, not compiled, not run on hardware
//
#include <hip/hip_runtime.h>

typedef __attribute__((ext_vector_type(16))) _Float16 v16h;
typedef __attribute__((ext_vector_type(8)))  float    v8f;

#define B_   128
#define N_   300
#define L_   600
#define KF   19
#define C1_  100
#define LC_  582
#define LP_  83
#define NC_  2
#define EPSF 1e-5f

#define KPAD 96     // padded K for both GEMMs (76->96, 83->96)
#define XSP  616    // padded positions staged in LDS (max read: 591+23=614)
#define PT   37     // position tiles of 16 (37*16=592 >= 582)
#define NT1  19     // filter tiles of 16 (19*16=304 >= 300)

__device__ inline v8f wmma_f16(v16h a, v16h b, v8f c) {
  return __builtin_amdgcn_wmma_f32_16x16x32_f16(false, a, false, b, (short)0, c,
                                                false, false);
}

// build a v16h fragment from two 8-half contiguous LDS runs
__device__ inline v16h ldsfrag2(const _Float16* p0, const _Float16* p1) {
  v16h v;
#pragma unroll
  for (int e = 0; e < 8; ++e) v[e] = p0[e];
#pragma unroll
  for (int e = 0; e < 8; ++e) v[e + 8] = p1[e];
  return v;
}

// -------------------------------------------------------------------------
// Kernel 1: conv(4ch,K=19) + bias + BN1 + exp  ->  maxpool(7,7)
// grid = (19 filter-tiles, 128 batches), block = 128 (4 waves)
// GEMM per tile: 16 positions x 16 filters, K=96 (3 WMMA k-steps)
// -------------------------------------------------------------------------
__global__ __launch_bounds__(128)
void k1_conv_pool(const float* __restrict__ x,  const float* __restrict__ cw,
                  const float* __restrict__ cb,
                  const float* __restrict__ g1, const float* __restrict__ b1,
                  const float* __restrict__ m1, const float* __restrict__ v1,
                  float* __restrict__ pooled) {
  __shared__ _Float16 xs[XSP * 4];      // x staged f16, [pos][ch] interleaved
  __shared__ _Float16 ws[16 * KPAD];    // weights f16, [filter][kk], kk=tap*4+ch
  __shared__ float alf[16], bet[16];    // folded BN1 scale/shift
  __shared__ float eb[16 * 592];        // exp(conv); slots >=582 are dead pad

  const int nt  = blockIdx.x;
  const int b   = blockIdx.y;
  const int tid = threadIdx.x;

  for (int i = tid; i < XSP * 4; i += 128) {
    int pos = i >> 2, c = i & 3;
    float v = (pos < L_) ? x[(b * 4 + c) * L_ + pos] : 0.0f;
    xs[i] = (_Float16)v;
  }
  for (int i = tid; i < 16 * KPAD; i += 128) {
    int n = i / KPAD, kk = i % KPAD;
    int tap = kk >> 2, c = kk & 3;
    int nf = nt * 16 + n;
    float v = (tap < KF && nf < N_) ? cw[(nf * 4 + c) * KF + tap] : 0.0f;
    ws[i] = (_Float16)v;
  }
  if (tid < 16) {
    int nf = nt * 16 + tid;
    if (nf < N_) {
      float sc = g1[nf] * rsqrtf(v1[nf] + EPSF);
      alf[tid] = sc;
      bet[tid] = sc * (cb[nf] - m1[nf]) + b1[nf];
    } else { alf[tid] = 0.0f; bet[tid] = 0.0f; }
  }
  __syncthreads();

  const int wave = tid >> 5;
  const int lane = tid & 31;
  const int hl   = lane >> 4;   // lane half
  const int col  = lane & 15;

  // hoist folded BN params into registers (one DS load each per wave)
  const float sa = alf[col];
  const float sb = bet[col];

  // B fragments: 16 filters (N=col), reused across all position tiles.
  // elem e of k-step s -> K = 32s + 8*hl + e (e<8), 32s + 16 + 8*hl + (e-8) (e>=8).
  const _Float16* wb = &ws[col * KPAD];
  v16h bf0 = ldsfrag2(wb + 0  + 8 * hl, wb + 16 + 8 * hl);
  v16h bf1 = ldsfrag2(wb + 32 + 8 * hl, wb + 48 + 8 * hl);
  v16h bf2 = ldsfrag2(wb + 64 + 8 * hl, wb + 80 + 8 * hl);

  for (int t = wave; t < PT; t += 4) {
    int l = t * 16 + col;                   // A row: M = lane%16
    const _Float16* ab = &xs[l * 4];        // xs[l*4 + kk] is contiguous in kk
    v16h a0 = ldsfrag2(ab + 0  + 8 * hl, ab + 16 + 8 * hl);
    v16h a1 = ldsfrag2(ab + 32 + 8 * hl, ab + 48 + 8 * hl);
    v16h a2 = ldsfrag2(ab + 64 + 8 * hl, ab + 80 + 8 * hl);
    v8f acc = {};
    acc = wmma_f16(a0, bf0, acc);
    acc = wmma_f16(a1, bf1, acc);
    acc = wmma_f16(a2, bf2, acc);
    // Branchless epilogue: D row r maps to position t*16 + 8*hl + r, so the
    // 8 outputs per lane are contiguous + 32B-aligned in eb. Positions
    // 582..591 land in the dead pad of eb and are never read by pooling.
    float4 o0, o1;
    o0.x = __expf(sa * acc[0] + sb);
    o0.y = __expf(sa * acc[1] + sb);
    o0.z = __expf(sa * acc[2] + sb);
    o0.w = __expf(sa * acc[3] + sb);
    o1.x = __expf(sa * acc[4] + sb);
    o1.y = __expf(sa * acc[5] + sb);
    o1.z = __expf(sa * acc[6] + sb);
    o1.w = __expf(sa * acc[7] + sb);
    float* dst = &eb[col * 592 + t * 16 + 8 * hl];
    *(float4*)(dst)     = o0;
    *(float4*)(dst + 4) = o1;
  }
  __syncthreads();

  for (int i = tid; i < 16 * LP_; i += 128) {
    int n = i / LP_, p = i % LP_;
    int nf = nt * 16 + n;
    if (nf < N_) {
      float mx = eb[n * 592 + p * 7];
#pragma unroll
      for (int j = 1; j < 7; ++j) mx = fmaxf(mx, eb[n * 592 + p * 7 + j]);
      pooled[(b * N_ + nf) * LP_ + p] = mx;
    }
  }
}

// -------------------------------------------------------------------------
// Kernel 2: per-unit FC1 (83->100) + BN2 + ReLU, then FC2 (100->1) + BN3 +
// ReLU. grid = 300 units, block = 256 (8 waves; wave = batch m-tile).
// GEMM: M=128 batch, N=112 (100 padded), K=96 (83 padded).
// -------------------------------------------------------------------------
__global__ __launch_bounds__(256)
void k2_fc(const float* __restrict__ pooled, const float* __restrict__ w1,
           const float* __restrict__ fb1,
           const float* __restrict__ g2, const float* __restrict__ b2p,
           const float* __restrict__ m2, const float* __restrict__ v2,
           const float* __restrict__ w2, const float* __restrict__ fb2,
           const float* __restrict__ g3, const float* __restrict__ b3,
           const float* __restrict__ m3, const float* __restrict__ v3,
           float* __restrict__ out2) {
  __shared__ _Float16 ps[B_ * KPAD];    // pooled activations f16 [batch][p]
  __shared__ _Float16 wt[112 * KPAD];   // fc1 weights f16 [h][p]
  __shared__ float a2[112], bb2[112];   // folded BN2 (fc1 bias included)

  const int n   = blockIdx.x;
  const int tid = threadIdx.x;

  for (int i = tid; i < B_ * KPAD; i += 256) {
    int bb = i / KPAD, p = i % KPAD;
    float v = (p < LP_) ? pooled[(bb * N_ + n) * LP_ + p] : 0.0f;
    ps[i] = (_Float16)v;
  }
  for (int i = tid; i < 112 * KPAD; i += 256) {
    int h = i / KPAD, p = i % KPAD;
    float v = (h < C1_ && p < LP_) ? w1[(n * C1_ + h) * LP_ + p] : 0.0f;
    wt[i] = (_Float16)v;
  }
  if (tid < 112) {
    int h = tid;
    if (h < C1_) {
      int idx = n * C1_ + h;
      float sc = g2[idx] * rsqrtf(v2[idx] + EPSF);
      a2[h]  = sc;
      bb2[h] = sc * (fb1[idx] - m2[idx]) + b2p[idx];
    } else { a2[h] = 0.0f; bb2[h] = 0.0f; }
  }
  __syncthreads();

  const int wave = tid >> 5;            // m-tile (16 batches)
  const int lane = tid & 31;
  const int hl   = lane >> 4;
  const int col  = lane & 15;

  const _Float16* ab = &ps[(wave * 16 + col) * KPAD];
  v16h A0 = ldsfrag2(ab + 0  + 8 * hl, ab + 16 + 8 * hl);
  v16h A1 = ldsfrag2(ab + 32 + 8 * hl, ab + 48 + 8 * hl);
  v16h A2 = ldsfrag2(ab + 64 + 8 * hl, ab + 80 + 8 * hl);

  float act[7][8];                      // BN2+ReLU activations, in registers
  float w2l[7];                         // fc2 weight for this lane's h columns
#pragma unroll
  for (int j = 0; j < 7; ++j) {
    int h = j * 16 + col;
    const _Float16* wb = &wt[h * KPAD];
    v16h B0 = ldsfrag2(wb + 0  + 8 * hl, wb + 16 + 8 * hl);
    v16h B1 = ldsfrag2(wb + 32 + 8 * hl, wb + 48 + 8 * hl);
    v16h B2 = ldsfrag2(wb + 64 + 8 * hl, wb + 80 + 8 * hl);
    v8f acc = {};
    acc = wmma_f16(A0, B0, acc);
    acc = wmma_f16(A1, B1, acc);
    acc = wmma_f16(A2, B2, acc);
    float sa = a2[h], sb = bb2[h];      // zero rows for h>=100 => act 0
#pragma unroll
    for (int r = 0; r < 8; ++r) act[j][r] = fmaxf(0.0f, sa * acc[r] + sb);
    w2l[j] = (h < C1_) ? w2[n * C1_ + h] : 0.0f;
  }

  // FC2: reduce over h = 7 tiles (registers) x 16 lane-columns (shfl butterfly)
  float part[8];
#pragma unroll
  for (int r = 0; r < 8; ++r) {
    float s = 0.0f;
#pragma unroll
    for (int j = 0; j < 7; ++j) s += act[j][r] * w2l[j];
#pragma unroll
    for (int msk = 1; msk < 16; msk <<= 1) s += __shfl_xor(s, msk, 32);
    part[r] = s;
  }
  if (col == 0) {
    float sc3 = g3[n] * rsqrtf(v3[n] + EPSF);
    float be3 = sc3 * (fb2[n] - m3[n]) + b3[n];
#pragma unroll
    for (int r = 0; r < 8; ++r) {
      int bi = wave * 16 + r + 8 * hl;  // D-row batch index
      out2[bi * N_ + n] = fmaxf(0.0f, sc3 * part[r] + be3);
    }
  }
}

// -------------------------------------------------------------------------
// Kernel 3: final 300 -> 2 linear. One block, thread = (batch, class).
// -------------------------------------------------------------------------
__global__ __launch_bounds__(256)
void k3_final(const float* __restrict__ out2, const float* __restrict__ fw,
              const float* __restrict__ fbf, float* __restrict__ out) {
  int tid = threadIdx.x;
  int b = tid >> 1, cls = tid & 1;
  float acc = fbf[cls];
  for (int nn = 0; nn < N_; ++nn)
    acc += out2[b * N_ + nn] * fw[cls * N_ + nn];
  out[b * NC_ + cls] = acc;
}

extern "C" void kernel_launch(void* const* d_in, const int* in_sizes, int n_in,
                              void* d_out, int out_size, void* d_ws,
                              size_t ws_size, hipStream_t stream) {
  (void)in_sizes; (void)n_in; (void)out_size; (void)ws_size;
  const float* x   = (const float*)d_in[0];
  const float* cw  = (const float*)d_in[1];
  const float* cb  = (const float*)d_in[2];
  const float* g1  = (const float*)d_in[3];
  const float* b1  = (const float*)d_in[4];
  const float* m1  = (const float*)d_in[5];
  const float* v1  = (const float*)d_in[6];
  const float* w1  = (const float*)d_in[7];
  const float* fb1 = (const float*)d_in[8];
  const float* g2  = (const float*)d_in[9];
  const float* b2  = (const float*)d_in[10];
  const float* m2  = (const float*)d_in[11];
  const float* v2  = (const float*)d_in[12];
  const float* w2  = (const float*)d_in[13];
  const float* fb2 = (const float*)d_in[14];
  const float* g3  = (const float*)d_in[15];
  const float* b3  = (const float*)d_in[16];
  const float* m3  = (const float*)d_in[17];
  const float* v3  = (const float*)d_in[18];
  const float* fw  = (const float*)d_in[19];
  const float* fbf = (const float*)d_in[20];

  float* pooled = (float*)d_ws;                       // [128][300][83] ~12.2MB
  float* o2     = pooled + (size_t)B_ * N_ * LP_;     // [128][300]

  k1_conv_pool<<<dim3(NT1, B_), 128, 0, stream>>>(x, cw, cb, g1, b1, m1, v1,
                                                  pooled);
  k2_fc<<<N_, 256, 0, stream>>>(pooled, w1, fb1, g2, b2, m2, v2, w2, fb2, g3,
                                b3, m3, v3, o2);
  k3_final<<<1, 256, 0, stream>>>(o2, fw, fbf, (float*)d_out);
}
